// MoE_72275709657594
// MI455X (gfx1250) — matrix operand order repaired and checked by reading, not compiled, not run on hardware
//
#include <hip/hip_runtime.h>
#include <hip/hip_bf16.h>

// Shapes from the reference
#define BATCH 4096
#define DIN   1024
#define DOUT  1024
#define NLEAF 32
#define KTOT  (DIN * NLEAF)   // 32768 : fused K dimension (k = i*32 + l)
#define NKCH  DIN             // 1024 k-chunks of 32 (one i per chunk)

typedef __attribute__((ext_vector_type(16))) __bf16 v16bf;
typedef __attribute__((ext_vector_type(8)))  float  v8f;

#define LDSTRIDE 40           // halfs per LDS tile row: 32 data + 8 pad (80B, 16B aligned, bank-conflict-free)

// ---------------- Kernel 1: gatings = softmax(x @ gw + gb, axis=1) ----------------
// One wave32 per batch row; lane == leaf index.
__global__ void gating_kernel(const float* __restrict__ x,
                              const float* __restrict__ gw,
                              const float* __restrict__ gb,
                              float* __restrict__ gout) {
  int row = blockIdx.x * (blockDim.x >> 5) + (threadIdx.x >> 5);
  int l   = threadIdx.x & 31;
  const float* xr = x + (size_t)row * DIN;
  float acc = gb[l];
  #pragma unroll 4
  for (int i = 0; i < DIN; i += 4) {
    float4 xv = *(const float4*)(xr + i);     // wave-uniform broadcast loads
    acc = fmaf(xv.x, gw[(i + 0) * NLEAF + l], acc);
    acc = fmaf(xv.y, gw[(i + 1) * NLEAF + l], acc);
    acc = fmaf(xv.z, gw[(i + 2) * NLEAF + l], acc);
    acc = fmaf(xv.w, gw[(i + 3) * NLEAF + l], acc);
  }
  float m = acc;
  for (int o = 16; o > 0; o >>= 1) m = fmaxf(m, __shfl_xor(m, o, 32));
  float e = __expf(acc - m);
  float s = e;
  for (int o = 16; o > 0; o >>= 1) s += __shfl_xor(s, o, 32);
  gout[(size_t)row * NLEAF + l] = e / s;
}

// ---------------- Kernel 2: xT[i][b] = x[b][i] (coalesced column access in GEMM) ----------------
__global__ void transpose_kernel(const float* __restrict__ x, float* __restrict__ xT) {
  __shared__ float tile[32][33];
  int bx = blockIdx.x * 32;            // i base
  int by = blockIdx.y * 32;            // b base
  int tx = threadIdx.x & 31;
  int ty = threadIdx.x >> 5;           // 0..7
  #pragma unroll
  for (int j = 0; j < 4; ++j)
    tile[ty + j * 8][tx] = x[(size_t)(by + ty + j * 8) * DIN + bx + tx];
  __syncthreads();
  #pragma unroll
  for (int j = 0; j < 4; ++j)
    xT[(size_t)(bx + ty + j * 8) * BATCH + by + tx] = tile[tx][ty + j * 8];
}

// ---------------- Kernel 3: pw fp32 -> bf16 (67 MB result is L2-resident, 192 MB L2) ----------------
__global__ void convert_pw_kernel(const float* __restrict__ pw, __bf16* __restrict__ out) {
  int idx = blockIdx.x * blockDim.x + threadIdx.x;      // one float4 per thread
  float4 v = ((const float4*)pw)[idx];
  union { __bf16 h[4]; uint2 u; } p;
  p.h[0] = (__bf16)v.x; p.h[1] = (__bf16)v.y;
  p.h[2] = (__bf16)v.z; p.h[3] = (__bf16)v.w;
  ((uint2*)out)[idx] = p.u;
}

// A-matrix 16x32 bf16 fragment (ISA 7.12.2): lane m<16 holds row m, K 0..7 (v0-3) and 16..23 (v4-7);
// lanes 16..31 hold K 8..15 and 24..31.
__device__ __forceinline__ v16bf load_fragA(const __bf16* base, int lane) {
  int r  = lane & 15;
  int hi = lane >> 4;
  const __bf16* p = base + r * LDSTRIDE + hi * 8;
  union { v16bf v; uint4 q[2]; } u;
  u.q[0] = *(const uint4*)(p);
  u.q[1] = *(const uint4*)(p + 16);
  return u.v;
}
// B-matrix 32x16 bf16 fragment: lane n<16 holds column n, K 0..7 and 16..23; lanes 16..31 hold K 8..15 and 24..31.
__device__ __forceinline__ v16bf load_fragB(const __bf16* base, int lane) {
  int c  = lane & 15;
  int hi = lane >> 4;
  const __bf16* p = base + c * LDSTRIDE + hi * 8;
  union { v16bf v; uint4 q[2]; } u;
  u.q[0] = *(const uint4*)(p);
  u.q[1] = *(const uint4*)(p + 16);
  return u.v;
}

// ---------------- Kernel 4: C[b,o] = sum_k A[b,k] P[o,k] + bias; A built on the fly ----------------
// Block tile 128x128, 8 waves in 2x4; wave tile 64x32 = 4x2 wmma accumulators. K-chunk = 32 (one i).
__global__ void __launch_bounds__(256)
moe_gemm_kernel(const float* __restrict__ xT,      // [DIN][BATCH]
                const __bf16* __restrict__ pwb,    // [DOUT][KTOT]
                const float* __restrict__ gws,     // [BATCH][NLEAF]
                const float* __restrict__ pb,      // [DOUT][NLEAF]
                float* __restrict__ out) {         // [BATCH][DOUT]
  __shared__ __align__(16) __bf16 Asm[2 * 128 * LDSTRIDE];   // 20 KB
  __shared__ __align__(16) __bf16 Bsm[2 * 128 * LDSTRIDE];   // 20 KB

  const int tid   = threadIdx.x;
  const int lane  = tid & 31;
  const int wid   = tid >> 5;        // 0..7
  const int wm    = wid & 1;         // wave M (64 rows each)
  const int wn    = wid >> 1;        // wave N (32 cols each)
  const int mbase = blockIdx.y * 128;
  const int nbase = blockIdx.x * 128;

  // fill assignment: thread t handles tile row t/2, leaf half (t&1)*16
  const int frow = tid >> 1;
  const int fl0  = (tid & 1) * 16;

  // gate values for this thread's A rows: constant across all 1024 k-chunks -> registers
  float greg[16];
  {
    const float* gp = gws + (size_t)(mbase + frow) * NLEAF + fl0;
    #pragma unroll
    for (int j = 0; j < 16; j += 4) {
      float4 t = *(const float4*)(gp + j);
      greg[j] = t.x; greg[j + 1] = t.y; greg[j + 2] = t.z; greg[j + 3] = t.w;
    }
  }
  const __bf16* pwrow = pwb + (size_t)(nbase + frow) * KTOT + fl0;

  v8f acc[4][2] = {};

  auto fill = [&](int kc, int buf) {
    // B tile: pw rows nbase..+128, k-slice kc*32..+32 (L2 hits after first M-tile pass)
    const uint4* src = (const uint4*)(pwrow + kc * NLEAF);
    uint4 b0 = src[0];
    uint4 b1 = src[1];
    // A tile on the fly: A[row, l] = xT[kc, mbase+row] * g[row, l]
    float xv = xT[(size_t)kc * BATCH + mbase + frow];
    union { __bf16 h[16]; uint4 q[2]; } pk;
    #pragma unroll
    for (int j = 0; j < 16; ++j) pk.h[j] = (__bf16)(xv * greg[j]);
    __bf16* brow = Bsm + buf * (128 * LDSTRIDE) + frow * LDSTRIDE + fl0;
    ((uint4*)brow)[0] = b0;
    ((uint4*)brow)[1] = b1;
    __bf16* arow = Asm + buf * (128 * LDSTRIDE) + frow * LDSTRIDE + fl0;
    ((uint4*)arow)[0] = pk.q[0];
    ((uint4*)arow)[1] = pk.q[1];
  };

  fill(0, 0);
  __syncthreads();

  #pragma unroll 1
  for (int kc = 0; kc < NKCH; ++kc) {
    const int cur = kc & 1;
    if (kc + 1 < NKCH) fill(kc + 1, cur ^ 1);     // prefetch into other buffer
    const __bf16* Ab = Asm + cur * (128 * LDSTRIDE);
    const __bf16* Bb = Bsm + cur * (128 * LDSTRIDE);
    v16bf bfrag[2];
    #pragma unroll
    for (int n = 0; n < 2; ++n)
      bfrag[n] = load_fragB(Bb + (wn * 32 + n * 16) * LDSTRIDE, lane);
    #pragma unroll
    for (int m = 0; m < 4; ++m) {
      v16bf afrag = load_fragA(Ab + (wm * 64 + m * 16) * LDSTRIDE, lane);
      #pragma unroll
      for (int n = 0; n < 2; ++n)
        acc[m][n] = __builtin_amdgcn_wmma_f32_16x16x32_bf16(
            false, afrag, false, bfrag[n], (short)0, acc[m][n], false, false);
    }
    __syncthreads();
  }

  // Epilogue: out[row,col] = acc + dot(g[row,:], pb[col,:])  (C layout: VGPR r -> M=r+8*(lane>=16), N=lane&15)
  const int rhi = lane >> 4;
  const int cl  = lane & 15;
  #pragma unroll
  for (int n = 0; n < 2; ++n) {
    const int col = nbase + wn * 32 + n * 16 + cl;
    const float* pbrow = pb + (size_t)col * NLEAF;
    float pbr[NLEAF];
    #pragma unroll
    for (int l = 0; l < NLEAF; l += 4) {
      float4 t = *(const float4*)(pbrow + l);
      pbr[l] = t.x; pbr[l + 1] = t.y; pbr[l + 2] = t.z; pbr[l + 3] = t.w;
    }
    #pragma unroll
    for (int m = 0; m < 4; ++m) {
      #pragma unroll
      for (int r = 0; r < 8; ++r) {
        const int row = mbase + wm * 64 + m * 16 + rhi * 8 + r;
        const float* grow = gws + (size_t)row * NLEAF;
        float bsum = 0.f;
        #pragma unroll
        for (int l = 0; l < NLEAF; l += 4) {
          float4 gv = *(const float4*)(grow + l);
          bsum = fmaf(gv.x, pbr[l], bsum);
          bsum = fmaf(gv.y, pbr[l + 1], bsum);
          bsum = fmaf(gv.z, pbr[l + 2], bsum);
          bsum = fmaf(gv.w, pbr[l + 3], bsum);
        }
        out[(size_t)row * DOUT + col] = acc[m][n][r] + bsum;
      }
    }
  }
}

extern "C" void kernel_launch(void* const* d_in, const int* in_sizes, int n_in,
                              void* d_out, int out_size, void* d_ws, size_t ws_size,
                              hipStream_t stream) {
  const float* x  = (const float*)d_in[0];   // [4096,1024]
  const float* gw = (const float*)d_in[1];   // [1024,32]
  const float* gb = (const float*)d_in[2];   // [32]
  const float* pw = (const float*)d_in[3];   // [1024,1024,32]
  const float* pb = (const float*)d_in[4];   // [1024,32]
  float* out = (float*)d_out;                // [4096,1024]

  // Workspace layout (needs ~84.4 MB): g fp32 0.5MB | xT fp32 16MB | pw bf16 64MB
  char* ws = (char*)d_ws;
  float*  g    = (float*)ws;
  float*  xT   = (float*)(ws + (512u << 10));
  __bf16* pwb  = (__bf16*)(ws + (512u << 10) + (16u << 20));

  gating_kernel<<<BATCH / 8, 256, 0, stream>>>(x, gw, gb, g);
  transpose_kernel<<<dim3(DIN / 32, BATCH / 32), 256, 0, stream>>>(x, xT);
  convert_pw_kernel<<<((size_t)DOUT * DIN * NLEAF / 4) / 256, 256, 0, stream>>>(pw, pwb);
  moe_gemm_kernel<<<dim3(DOUT / 128, BATCH / 128), 256, 0, stream>>>(xT, pwb, g, pb, out);
}
